// IDM_SGC_52733608461002
// MI455X (gfx1250) — compile-verified, matrix operand order checked by reading, not run on hardware
//
#include <hip/hip_runtime.h>

// CDNA5 / gfx1250, wave32. WMMA vector types.
typedef __attribute__((ext_vector_type(16))) __bf16 v16bf;
typedef __attribute__((ext_vector_type(8)))  float  v8f;
typedef __attribute__((ext_vector_type(2)))  float  v2f;

#define M_DIM 128
#define N_DIM 8192
#define K_DIM 8192

// ---- bf16 split helpers (round-to-nearest-even) ----
static __device__ __forceinline__ unsigned short f2bf(float x) {
  unsigned int u = __float_as_uint(x);
  u += 0x7FFFu + ((u >> 16) & 1u);
  return (unsigned short)(u >> 16);
}
static __device__ __forceinline__ float bf2f(unsigned short b) {
  return __uint_as_float(((unsigned int)b) << 16);
}

// ---- CDNA5 async Global->LDS DMA (tracked by ASYNCcnt) ----
// Generic pointers to __shared__ carry the LDS byte offset in their low 32 bits
// (ISA: LDS aperture addressing).
static __device__ __forceinline__ void async_load_b128(void* lds_dst, const void* gsrc) {
  unsigned lds = (unsigned)(uintptr_t)lds_dst;
  asm volatile("global_load_async_to_lds_b128 %0, %1, off"
               :: "v"(lds), "v"(gsrc)
               : "memory");
}
static __device__ __forceinline__ void wait_asynccnt0() {
  asm volatile("s_wait_asynccnt 0" ::: "memory");
}

// ---------- FF = F^T F (128x128, trivial cost) ----------
__global__ __launch_bounds__(256) void k_ftf(const float* __restrict__ F,
                                             float* __restrict__ FF) {
  int idx = blockIdx.x * 256 + threadIdx.x;   // 16384 outputs
  int i = idx >> 7, j = idx & 127;
  float s = 0.f;
#pragma unroll 4
  for (int k = 0; k < 128; ++k) s = fmaf(F[k * 128 + i], F[k * 128 + j], s);
  FF[idx] = s;
}

// ---------- A0 = gamma * FF / (||FF||_F + 1e-6), deterministic tree reduce ----------
__global__ __launch_bounds__(256) void k_norm_scale(const float* __restrict__ FF,
                                                    const float* __restrict__ gamma,
                                                    float* __restrict__ A0) {
  __shared__ float red[256];
  int t = threadIdx.x;
  float s = 0.f;
  for (int i = t; i < 16384; i += 256) s = fmaf(FF[i], FF[i], s);
  red[t] = s; __syncthreads();
  for (int off = 128; off > 0; off >>= 1) {
    if (t < off) red[t] += red[t + off];
    __syncthreads();
  }
  float scale = gamma[0] / (sqrtf(red[0]) + 1e-6f);
  for (int i = t; i < 16384; i += 256) A0[i] = FF[i] * scale;
}

__global__ void k_copy(const float* __restrict__ src, float* __restrict__ dst, int n) {
  int i = blockIdx.x * 256 + threadIdx.x;
  if (i < n) dst[i] = src[i];
}
__global__ void k_square(float* __restrict__ d, int n) {
  int i = blockIdx.x * 256 + threadIdx.x;
  if (i < n) d[i] *= d[i];
}
// f32 -> (hi, lo) bf16 decomposition for split-precision GEMM
__global__ void k_split(const float* __restrict__ x, unsigned short* __restrict__ hi,
                        unsigned short* __restrict__ lo, int n) {
  int i = blockIdx.x * 256 + threadIdx.x;
  if (i < n) {
    float v = x[i];
    unsigned short h = f2bf(v);
    hi[i] = h;
    lo[i] = f2bf(v - bf2f(h));
  }
}
// S[i,j] += T[i,j] * d[j]   (float4 vectorized, row width 8192 -> 2048 chunks)
__global__ __launch_bounds__(256) void k_update(float* __restrict__ S,
                                                const float* __restrict__ T,
                                                const float* __restrict__ d) {
  int i = blockIdx.x * 256 + threadIdx.x;          // 262144 float4 total
  float4 s = ((const float4*)S)[i];
  float4 t4 = ((const float4*)T)[i];
  float4 dv = ((const float4*)d)[i & 2047];
  s.x = fmaf(t4.x, dv.x, s.x);
  s.y = fmaf(t4.y, dv.y, s.y);
  s.z = fmaf(t4.z, dv.z, s.z);
  s.w = fmaf(t4.w, dv.w, s.w);
  ((float4*)S)[i] = s;
}

// ---------- Big split-bf16 GEMM: C(128xN) = A(128x8192) @ B or B^T ----------
// A pre-split bf16 hi/lo async-DMA'd straight into padded LDS; B (Q_S, f32)
// async-DMA'd into an LDS staging buffer, then split to bf16 hi/lo in LDS.
// Double-buffered: step k+1 DMAs overlap step k WMMA. 256 thr = 8 waves,
// tile 128(M) x 32(N), K staged 32 at a time. LDS rows padded to 34 bf16
// (17 banks, coprime with 64) -> conflict-free fragment gathers.
template <bool TRANSB>
__global__ __launch_bounds__(256) void big_gemm_split(
    const unsigned short* __restrict__ Ahi, const unsigned short* __restrict__ Alo,
    const float* __restrict__ B, float* __restrict__ C) {
  __shared__ unsigned short sAhi[2][128 * 34];
  __shared__ unsigned short sAlo[2][128 * 34];
  __shared__ float          sBf [2][32 * 32];
  __shared__ unsigned short sBhi[32 * 34];
  __shared__ unsigned short sBlo[32 * 34];

  const int tid = threadIdx.x;
  const int w = tid >> 5, L = tid & 31, h = L >> 4, lm = L & 15;
  const int n0 = blockIdx.x * 32;

  // per-thread DMA coordinates
  const int ar = tid >> 1, aseg = (tid & 1) * 16;  // A: 16 bf16 = 2 x b128 per matrix
  const int bkr = tid >> 3, bc4 = (tid & 7) * 4;   // B: one b128 (4 f32)

  v8f c0 = {}; v8f c1 = {};

  // ---- async DMA issue helpers (5 x b128 per thread per K-step) ----
  auto issue_tiles = [&](int buf, int k0) {
    const unsigned short* gh = Ahi + (size_t)ar * K_DIM + k0 + aseg;
    const unsigned short* gl = Alo + (size_t)ar * K_DIM + k0 + aseg;
    unsigned short* dh = &sAhi[buf][ar * 34 + aseg];
    unsigned short* dl = &sAlo[buf][ar * 34 + aseg];
    async_load_b128(dh,     gh);
    async_load_b128(dh + 8, gh + 8);
    async_load_b128(dl,     gl);
    async_load_b128(dl + 8, gl + 8);
    const float* gb = TRANSB ? (B + (size_t)(n0 + bkr) * K_DIM + k0 + bc4)   // staging [n][k]
                             : (B + (size_t)(k0 + bkr) * N_DIM + n0 + bc4);  // staging [k][n]
    async_load_b128(&sBf[buf][bkr * 32 + bc4], gb);
  };

  // prologue: stage K-step 0
  issue_tiles(0, 0);
  wait_asynccnt0();
  __syncthreads();

  int buf = 0;
  for (int k0 = 0; k0 < K_DIM; k0 += 32) {
    // 1) kick off DMA for the next K-step (overlaps with convert+WMMA below)
    if (k0 + 32 < K_DIM) issue_tiles(buf ^ 1, k0 + 32);

    // 2) split current B staging tile f32 -> bf16 hi/lo into padded [n][k] layout
#pragma unroll
    for (int i = 0; i < 4; ++i) {
      float v = sBf[buf][bkr * 32 + bc4 + i];
      unsigned short hh = f2bf(v);
      unsigned short ll = f2bf(v - bf2f(hh));
      if (!TRANSB) { sBhi[(bc4 + i) * 34 + bkr] = hh; sBlo[(bc4 + i) * 34 + bkr] = ll; }
      else         { sBhi[bkr * 34 + bc4 + i] = hh; sBlo[bkr * 34 + bc4 + i] = ll; }
    }
    __syncthreads();

    // 3) fragments per documented 16-bit A/B layout:
    //    lane<16: K 0..7 (V0-3), 16..23 (V4-7); lane>=16: +8. K-pairs per VGPR.
    union Frag { unsigned int u[8]; v16bf v; };
    Frag ah, al;
    const int row = w * 16 + lm;
#pragma unroll
    for (int j = 0; j < 8; ++j) {
      int kk = ((j & 4) << 2) + h * 8 + ((j & 3) << 1);
      ah.u[j] = *(const unsigned int*)&sAhi[buf][row * 34 + kk];
      al.u[j] = *(const unsigned int*)&sAlo[buf][row * 34 + kk];
    }
#pragma unroll
    for (int nt = 0; nt < 2; ++nt) {
      Frag bh, bl;
      const int nn = nt * 16 + lm;
#pragma unroll
      for (int j = 0; j < 8; ++j) {
        int kk = ((j & 4) << 2) + h * 8 + ((j & 3) << 1);
        bh.u[j] = *(const unsigned int*)&sBhi[nn * 34 + kk];
        bl.u[j] = *(const unsigned int*)&sBlo[nn * 34 + kk];
      }
      v8f& cc = nt ? c1 : c0;
      // split product: hi*hi + hi*lo + lo*hi (lo*lo below f32 noise)
      cc = __builtin_amdgcn_wmma_f32_16x16x32_bf16(false, ah.v, false, bh.v, (short)0, cc, false, false);
      cc = __builtin_amdgcn_wmma_f32_16x16x32_bf16(false, ah.v, false, bl.v, (short)0, cc, false, false);
      cc = __builtin_amdgcn_wmma_f32_16x16x32_bf16(false, al.v, false, bh.v, (short)0, cc, false, false);
    }

    // 4) next-step DMA must have landed; barrier also protects sBhi/sBlo reuse
    wait_asynccnt0();
    __syncthreads();
    buf ^= 1;
  }

  // C layout: VGPR r -> M = r + 8*(lane>=16), N = lane&15
#pragma unroll
  for (int nt = 0; nt < 2; ++nt) {
    const v8f& cc = nt ? c1 : c0;
#pragma unroll
    for (int r = 0; r < 8; ++r) {
      int rowg = w * 16 + r + 8 * h;
      int colg = n0 + nt * 16 + lm;
      C[(size_t)rowg * N_DIM + colg] = cc[r];
    }
  }
}

// ---------- Small full-precision GEMM: C(128xN) = A(128x128) @ B(128xN) ----------
// Uses V_WMMA_F32_16X16X4_F32; K=128 only, loads straight from L2.
__global__ __launch_bounds__(256) void small_gemm_f32(const float* __restrict__ A,
                                                      const float* __restrict__ B,
                                                      float* __restrict__ C, int N) {
  const int w = threadIdx.x >> 5, L = threadIdx.x & 31, h = L >> 4, lm = L & 15;
  const int ntiles = N >> 4;
  const int tileId = blockIdx.x * 8 + w;
  const int tm = tileId / ntiles, tn = tileId % ntiles;
  v8f c = {};
  for (int kk = 0; kk < 32; ++kk) {
    // f32 16x4 A layout: lane<16 -> K = j (VGPR j); lane>=16 -> K = j+2
    int k = kk * 4 + 2 * h;
    v2f a, b;
    a.x = A[(tm * 16 + lm) * 128 + k];
    a.y = A[(tm * 16 + lm) * 128 + k + 1];
    b.x = B[(size_t)k * N + tn * 16 + lm];
    b.y = B[(size_t)(k + 1) * N + tn * 16 + lm];
    c = __builtin_amdgcn_wmma_f32_16x16x4_f32(false, a, false, b, (short)0, c, false, false);
  }
#pragma unroll
  for (int r = 0; r < 8; ++r)
    C[(size_t)(tm * 16 + r + 8 * h) * N + tn * 16 + lm] = c[r];
}

extern "C" void kernel_launch(void* const* d_in, const int* in_sizes, int n_in,
                              void* d_out, int out_size, void* d_ws, size_t ws_size,
                              hipStream_t stream) {
  const float* X     = (const float*)d_in[0];   // 128 x 8192
  const float* F     = (const float*)d_in[1];   // 128 x 128
  const float* Q_S   = (const float*)d_in[2];   // 8192 x 8192
  const float* LamS  = (const float*)d_in[3];   // 8192
  const float* gamma = (const float*)d_in[4];   // scalar
  float* Z = (float*)d_out;                     // 128 x 8192

  // workspace layout (floats): ~8.3 MB total
  float* ws = (float*)d_ws;
  float* FF = ws;                // 16384
  float* A0 = ws + 16384;        // 16384
  float* A1 = ws + 32768;        // 16384
  float* dv = ws + 49152;        // 8192
  float* S  = ws + 65536;        // 1,048,576  (Xh, then series accumulator)
  float* T  = S + 1048576;       // 1,048,576  (series temp; overlaid by bf16 hi/lo)
  unsigned short* Ahi = (unsigned short*)T;      // 1,048,576 bf16
  unsigned short* Alo = Ahi + 1048576;           // 1,048,576 bf16

  // 1) A0 = gamma * g(F) = gamma * F^T F / (||F^T F||_F + 1e-6)
  k_ftf<<<64, 256, 0, stream>>>(F, FF);
  k_norm_scale<<<1, 256, 0, stream>>>(FF, gamma, A0);
  k_copy<<<32, 256, 0, stream>>>(LamS, dv, 8192);

  // 2) S = Xh = X @ Q_S   (split-bf16 WMMA, async-DMA pipelined, ~256 MB read)
  k_split<<<4096, 256, 0, stream>>>(X, Ahi, Alo, 128 * 8192);
  big_gemm_split<false><<<256, 256, 0, stream>>>(Ahi, Alo, Q_S, S);

  // 3) Neumann doubling: S <- S + A^K @ S * d^K ; A <- A^2 ; d <- d^2
  //    6 doublings = 64 exact series terms (spectral radius ~0.2 -> exact to f32)
  float* Acur = A0;
  float* Anxt = A1;
  for (int it = 0; it < 6; ++it) {
    small_gemm_f32<<<512, 256, 0, stream>>>(Acur, S, T, 8192);      // T = A @ S
    small_gemm_f32<<<8,   256, 0, stream>>>(Acur, Acur, Anxt, 128); // A' = A @ A
    k_update<<<1024, 256, 0, stream>>>(S, T, dv);                   // S += T * d[col]
    k_square<<<32, 256, 0, stream>>>(dv, 8192);                     // d *= d
    float* tmp = Acur; Acur = Anxt; Anxt = tmp;
  }

  // 4) Z = S @ Q_S^T   (split-bf16 WMMA, transposed-B async tile path)
  k_split<<<4096, 256, 0, stream>>>(S, Ahi, Alo, 128 * 8192);
  big_gemm_split<true><<<256, 256, 0, stream>>>(Ahi, Alo, Q_S, Z);
}
